// Block_35424890257432
// MI455X (gfx1250) — matrix operand order repaired and checked by reading, not compiled, not run on hardware
//
#include <hip/hip_runtime.h>
#include <hip/hip_bf16.h>
#include <math.h>

// Problem constants (from reference)
constexpr int kNE   = 1024;
constexpr int kNH   = 16;
constexpr int kDH   = 64;    // kNE / kNH
constexpr int kNHID = 4096;
constexpr int kSL   = 2048;
constexpr int kBS   = 4;
constexpr int kTOK  = kBS * kSL;       // 8192 rows
constexpr float kEPS = 1e-5f;

typedef __attribute__((ext_vector_type(16))) _Float16 v16h;
typedef __attribute__((ext_vector_type(8)))  _Float16 v8h;
typedef __attribute__((ext_vector_type(4)))  _Float16 v4h;
typedef __attribute__((ext_vector_type(8)))  float    v8f;

static __device__ __forceinline__ v8f wmma_f16(v16h a, v16h b, v8f c) {
  // (neg_a, A, neg_b, B, c_mod, C, reuse_a, reuse_b)
  return __builtin_amdgcn_wmma_f32_16x16x32_f16(false, a, false, b, (short)0, c,
                                                false, false);
}

// CDNA5 async copy: global memory -> LDS, 16 bytes per lane, ASYNCcnt-tracked.
static __device__ __forceinline__ void async_copy_b128(unsigned lds_off,
                                                       const void* gptr) {
  asm volatile("global_load_async_to_lds_b128 %0, %1, off"
               :: "v"(lds_off), "v"((unsigned long long)(size_t)gptr)
               : "memory");
}

static __device__ __forceinline__ v16h cat8(v8h lo, v8h hi) {
  return __builtin_shufflevector(lo, hi, 0, 1, 2, 3, 4, 5, 6, 7,
                                 8, 9, 10, 11, 12, 13, 14, 15);
}

// A-fragment 16x32 f16 K pattern (ISA 7.12.2): halves 0..7 = K[koff..koff+7],
// halves 8..15 = K[16+koff..16+koff+7], koff = (lane&16) ? 8 : 0.

// ---------------------------------------------------------------------------
// f32 -> f16 bulk conversion (weights once; x once). n must be multiple of 4.
// ---------------------------------------------------------------------------
__global__ __launch_bounds__(256) void f32_to_f16_kernel(
    const float* __restrict__ src, _Float16* __restrict__ dst, int n)
{
  const int idx = (blockIdx.x * 256 + threadIdx.x) * 4;
  if (idx < n) {
    const float4 v = *(const float4*)(src + idx);
    v4h o;
    o[0] = (_Float16)v.x; o[1] = (_Float16)v.y;
    o[2] = (_Float16)v.z; o[3] = (_Float16)v.w;
    *(v4h*)(dst + idx) = o;
  }
}

// ---------------------------------------------------------------------------
// WMMA GEMM, f16 operands, fp32 accumulate, double-buffered async LDS staging:
//   C[M,N] = A[M,K] * B[K,N] + bias  (RELU / OUTH are compile-time)
// Block = 128 threads = 4 waves; block tile 128x128; wave tile 64x64.
// ---------------------------------------------------------------------------
constexpr int kAStr = 40;    // f16 per A-tile row (32 data + 8 pad)
constexpr int kBStr = 136;   // f16 per B-tile row (128 data + 8 pad)

template <int RELU, int OUTH>
__global__ __launch_bounds__(128) void wmma_gemm_f16(
    const _Float16* __restrict__ A, const _Float16* __restrict__ B,
    const float* __restrict__ bias, float* __restrict__ Cf,
    _Float16* __restrict__ Ch, int M, int N, int K)
{
  __shared__ __align__(16) _Float16 As[2][128 * kAStr];  // 2 x 10240 B
  __shared__ __align__(16) _Float16 Bs[2][32 * kBStr];   // 2 x  8704 B

  const int tid  = threadIdx.x;
  const int lane = tid & 31;
  const int wave = tid >> 5;                       // 0..3
  const int brow = blockIdx.y * 128;
  const int bcol = blockIdx.x * 128;
  const int wrow = (wave >> 1) * 64;
  const int wcol = (wave & 1) * 64;
  const int mloc = lane & 15;
  const int koff = (lane & 16) ? 8 : 0;

  // Issue 8 async b128 copies (per lane) staging a 128x32 A tile + 32x128 B tile.
  auto stage = [&](int k0, int buf) {
    const unsigned ab = (unsigned)(size_t)&As[buf][0];
    const unsigned bb = (unsigned)(size_t)&Bs[buf][0];
    #pragma unroll
    for (int i = 0; i < 4; ++i) {
      const int id = i * 128 + tid;
      const int r = id >> 2, c = id & 3;           // A row, 16B chunk
      async_copy_b128(ab + (unsigned)(r * kAStr + c * 8) * 2,
                      A + (size_t)(brow + r) * K + k0 + c * 8);
    }
    #pragma unroll
    for (int i = 0; i < 4; ++i) {
      const int id = i * 128 + tid;
      const int r = id >> 4, c = id & 15;          // B row, 16B chunk
      async_copy_b128(bb + (unsigned)(r * kBStr + c * 8) * 2,
                      B + (size_t)(k0 + r) * N + bcol + c * 8);
    }
  };

  const v8f vzero = {0.f, 0.f, 0.f, 0.f, 0.f, 0.f, 0.f, 0.f};
  v8f acc[4][4];
  #pragma unroll
  for (int i = 0; i < 4; ++i)
    #pragma unroll
    for (int j = 0; j < 4; ++j) acc[i][j] = vzero;

  stage(0, 0);
  const int nk = K >> 5;
  for (int t = 0; t < nk; ++t) {
    const int buf = t & 1;
    if (t + 1 < nk) {
      stage((t + 1) * 32, buf ^ 1);      // prefetch next tile behind compute
      asm volatile("s_wait_asynccnt 0x8" ::: "memory");  // tile t resident
    } else {
      asm volatile("s_wait_asynccnt 0x0" ::: "memory");
    }
    __syncthreads();

    v16h af[4], bf[4];
    #pragma unroll
    for (int i = 0; i < 4; ++i) {
      const int row = wrow + 16 * i + mloc;
      const v8h lo = *(const v8h*)&As[buf][row * kAStr + koff];
      const v8h hi = *(const v8h*)&As[buf][row * kAStr + 16 + koff];
      af[i] = cat8(lo, hi);
    }
    #pragma unroll
    for (int j = 0; j < 4; ++j) {
      const int col = wcol + 16 * j + mloc;        // B-frag: N = lane%16, K = A-frag pattern
      #pragma unroll
      for (int s = 0; s < 16; ++s)
        bf[j][s] = Bs[buf][((s < 8) ? koff + s : 8 + koff + s) * kBStr + col];
    }
    #pragma unroll
    for (int i = 0; i < 4; ++i)
      #pragma unroll
      for (int j = 0; j < 4; ++j)
        acc[i][j] = wmma_f16(af[i], bf[j], acc[i][j]);

    __syncthreads();   // readers done before next iteration's async writes
  }

  // ---- branch-free epilogue (RELU/OUTH compile-time) ----
  const int mbase = (lane & 16) ? 8 : 0;
  #pragma unroll
  for (int i = 0; i < 4; ++i) {
    const int row = brow + wrow + 16 * i + mbase;
    #pragma unroll
    for (int j = 0; j < 4; ++j) {
      const int n = bcol + wcol + 16 * j + mloc;
      const float bv = bias[n];
      #pragma unroll
      for (int r = 0; r < 8; ++r) {
        float v = acc[i][j][r] + bv;
        if (RELU) v = fmaxf(v, 0.f);
        const size_t off = (size_t)(row + r) * N + n;
        if (OUTH) Ch[off] = (_Float16)v;
        else      Cf[off] = v;
      }
    }
  }
}

// ---------------------------------------------------------------------------
// K-transpose: qkv K section -> kT[b][h][d][s] so attention's K^T fragments
// are loaded with consecutive s across lanes. LDS-tiled for coalescing both sides.
// ---------------------------------------------------------------------------
__global__ __launch_bounds__(256) void k_transpose_kernel(
    const _Float16* __restrict__ qkvh, _Float16* __restrict__ kT)
{
  __shared__ __align__(16) _Float16 tile[64][72];  // 64s x 64d (+8 pad)
  const int blk = blockIdx.x;
  const int st = blk & (kSL / 64 - 1);             // s tile
  const int bh = blk >> 5;
  const int h = bh & (kNH - 1);
  const int b = bh >> 4;
  const int s0 = st * 64;
  const int tid = threadIdx.x;
  {
    const int r = tid >> 2, c = (tid & 3) * 16;    // r = s offset, c = d chunk
    const _Float16* src =
        qkvh + (size_t)(b * kSL + s0 + r) * (3 * kNE) + kNE + h * kDH + c;
    *(v8h*)&tile[r][c]     = *(const v8h*)src;
    *(v8h*)&tile[r][c + 8] = *(const v8h*)(src + 8);
  }
  __syncthreads();
  {
    const int d = tid >> 2, c = (tid & 3) * 16;    // d row out, c = s chunk
    _Float16* dst = kT + ((size_t)(b * kNH + h) * kDH + d) * kSL + s0 + c;
    v8h a, bb;
    #pragma unroll
    for (int u = 0; u < 8; ++u) {
      a[u]  = tile[c + u][d];
      bb[u] = tile[c + 8 + u][d];
    }
    *(v8h*)dst       = a;
    *(v8h*)(dst + 8) = bb;
  }
}

// ---------------------------------------------------------------------------
// Flash attention (causal), f16 qkv/kT in, f16 y out, fp32 softmax+accumulate.
// One wave per 16-query tile per (b,h).
// ---------------------------------------------------------------------------
__global__ __launch_bounds__(32) void flash_attn_kernel(
    const _Float16* __restrict__ qkv, const _Float16* __restrict__ kT,
    _Float16* __restrict__ y)
{
  __shared__ __align__(16) _Float16 pt[16][32];   // P: C-layout -> A-layout

  const int lane = threadIdx.x;
  const int tile = blockIdx.x;
  const int qt = tile & (kSL / 16 - 1);
  const int bh = tile >> 7;
  const int h = bh & (kNH - 1);
  const int b = bh >> 4;
  const int q0 = qt * 16;
  const size_t rs = (size_t)(3 * kNE);            // 3072 f16 per token row

  const _Float16* base = qkv + (size_t)b * kSL * rs;
  const _Float16* Qp = base + h * kDH;
  const _Float16* Vp = base + 2 * kNE + h * kDH;
  const _Float16* Kt = kT + (size_t)(b * kNH + h) * kDH * kSL;  // [d][s]

  const int mloc = lane & 15;
  const int koff = (lane & 16) ? 8 : 0;
  const int rbase = (lane & 16) ? 8 : 0;

  // Q fragments (16x32 f16) for d in [0,32) and [32,64): two b128 loads each.
  v16h qa[2];
  {
    const _Float16* qr = Qp + (size_t)(q0 + mloc) * rs;
    #pragma unroll
    for (int half = 0; half < 2; ++half) {
      const v8h lo = *(const v8h*)(qr + 32 * half + koff);
      const v8h hi = *(const v8h*)(qr + 32 * half + 16 + koff);
      qa[half] = cat8(lo, hi);
    }
  }

  const v8f vzero = {0.f, 0.f, 0.f, 0.f, 0.f, 0.f, 0.f, 0.f};
  float rowm[8], rowl[8];
  v8f o[4];
  #pragma unroll
  for (int r = 0; r < 8; ++r) { rowm[r] = -__builtin_inff(); rowl[r] = 0.f; }
  #pragma unroll
  for (int t = 0; t < 4; ++t) o[t] = vzero;

  const int nkb = (q0 + 16 + 31) >> 5;
  for (int kb = 0; kb < nkb; ++kb) {
    const int j0 = kb * 32;

    // S tile: 16 queries x 32 keys (two 16x16 fragments); K^T frag = [d=A-frag K pattern][s=lane%16]
    v8f s0 = vzero, s1 = vzero;
    #pragma unroll
    for (int ch = 0; ch < 2; ++ch) {
      const _Float16* kr = Kt + (size_t)(32 * ch) * kSL + j0 + mloc;
      v16h kf0, kf1;
      #pragma unroll
      for (int s = 0; s < 16; ++s) {
        const int kk = (s < 8) ? koff + s : 8 + koff + s;
        kf0[s] = kr[(size_t)kk * kSL];
        kf1[s] = kr[(size_t)kk * kSL + 16];
      }
      s0 = wmma_f16(qa[ch], kf0, s0);
      s1 = wmma_f16(qa[ch], kf1, s1);
    }

    // Online softmax (row r lives across a 16-lane half-wave); 1/sqrt(64) here.
    #pragma unroll
    for (int r = 0; r < 8; ++r) {
      const int qrow = q0 + rbase + r;
      float v0 = (j0 + mloc      <= qrow) ? s0[r] * 0.125f : -__builtin_inff();
      float v1 = (j0 + 16 + mloc <= qrow) ? s1[r] * 0.125f : -__builtin_inff();
      float mx = fmaxf(v0, v1);
      #pragma unroll
      for (int ofs = 1; ofs < 16; ofs <<= 1) mx = fmaxf(mx, __shfl_xor(mx, ofs, 32));
      const float mnew = fmaxf(rowm[r], mx);
      const float alpha = __expf(rowm[r] - mnew);
      const float p0 = __expf(v0 - mnew);
      const float p1 = __expf(v1 - mnew);
      float psum = p0 + p1;
      #pragma unroll
      for (int ofs = 1; ofs < 16; ofs <<= 1) psum += __shfl_xor(psum, ofs, 32);
      rowl[r] = rowl[r] * alpha + psum;
      rowm[r] = mnew;
      #pragma unroll
      for (int t = 0; t < 4; ++t) o[t][r] = o[t][r] * alpha;
      pt[rbase + r][mloc]      = (_Float16)p0;
      pt[rbase + r][16 + mloc] = (_Float16)p1;
    }
    __syncthreads();

    // P as A-fragment (two b128 ds loads)
    const v8h plo = *(const v8h*)&pt[mloc][koff];
    const v8h phi = *(const v8h*)&pt[mloc][16 + koff];
    const v16h pa = cat8(plo, phi);
    __syncthreads();

    // O += P * V
    #pragma unroll
    for (int t = 0; t < 4; ++t) {
      const _Float16* vr = Vp + (size_t)j0 * rs + 16 * t + mloc;  // N = lane%16, K = A-frag pattern
      v16h vf;
      #pragma unroll
      for (int s = 0; s < 16; ++s)
        vf[s] = vr[(size_t)((s < 8) ? koff + s : 8 + koff + s) * rs];
      o[t] = wmma_f16(pa, vf, o[t]);
    }
  }

  // Epilogue: normalize and store y (f16) at [b,s, h*64 + d]
  #pragma unroll
  for (int r = 0; r < 8; ++r) {
    const float inv = 1.0f / rowl[r];
    #pragma unroll
    for (int t = 0; t < 4; ++t) {
      y[((size_t)b * kSL + q0 + rbase + r) * kNE + h * kDH + 16 * t + mloc] =
          (_Float16)(o[t][r] * inv);
    }
  }
}

// ---------------------------------------------------------------------------
// Fused residual + LayerNorm (unbiased std, eps added to std — matches ref).
// Emits f32 (residual path) and optionally f16 (next GEMM's A operand).
// ---------------------------------------------------------------------------
static __device__ __forceinline__ float block_reduce_sum(float v, float* red) {
  const int tid = threadIdx.x;
  #pragma unroll
  for (int o = 1; o < 32; o <<= 1) v += __shfl_xor(v, o, 32);
  if ((tid & 31) == 0) red[tid >> 5] = v;
  __syncthreads();
  if (tid < 32) {
    float t = (tid < 8) ? red[tid] : 0.f;
    #pragma unroll
    for (int o = 1; o < 8; o <<= 1) t += __shfl_xor(t, o, 32);
    if (tid == 0) red[0] = t;
  }
  __syncthreads();
  const float out = red[0];
  __syncthreads();
  return out;
}

template <int EMIT_HALF>
__global__ __launch_bounds__(256) void add_ln_kernel(
    const float* __restrict__ x, const float* __restrict__ resid,
    const float* __restrict__ gamma, const float* __restrict__ beta,
    float* __restrict__ outf, _Float16* __restrict__ outh)
{
  __shared__ float red[8];
  const int row = blockIdx.x;
  const int tid = threadIdx.x;
  const float* xr = x + (size_t)row * kNE;
  const float* rr = resid + (size_t)row * kNE;

  float vals[4];
  float s = 0.f;
  #pragma unroll
  for (int i = 0; i < 4; ++i) {
    const int idx = tid + 256 * i;
    vals[i] = xr[idx] + rr[idx];
    s += vals[i];
  }
  const float mean = block_reduce_sum(s, red) * (1.0f / kNE);

  float ss = 0.f;
  #pragma unroll
  for (int i = 0; i < 4; ++i) { const float d = vals[i] - mean; ss += d * d; }
  const float var = block_reduce_sum(ss, red) * (1.0f / (kNE - 1));  // ddof=1
  const float inv = 1.0f / (sqrtf(var) + kEPS);                      // eps on std

  #pragma unroll
  for (int i = 0; i < 4; ++i) {
    const int idx = tid + 256 * i;
    const float v = (vals[i] - mean) * inv * gamma[idx] + beta[idx];
    outf[(size_t)row * kNE + idx] = v;
    if (EMIT_HALF) outh[(size_t)row * kNE + idx] = (_Float16)v;
  }
}

// ---------------------------------------------------------------------------
extern "C" void kernel_launch(void* const* d_in, const int* in_sizes, int n_in,
                              void* d_out, int out_size, void* d_ws, size_t ws_size,
                              hipStream_t stream) {
  const float* x     = (const float*)d_in[0];
  // d_in[1] = causal mask (handled analytically; unused)
  const float* qkv_w = (const float*)d_in[2];
  const float* qkv_b = (const float*)d_in[3];
  const float* o_w   = (const float*)d_in[4];
  const float* o_b   = (const float*)d_in[5];
  const float* ln1_a = (const float*)d_in[6];
  const float* ln1_b = (const float*)d_in[7];
  const float* w1    = (const float*)d_in[8];
  const float* b1    = (const float*)d_in[9];
  const float* w2    = (const float*)d_in[10];
  const float* b2    = (const float*)d_in[11];
  const float* ln2_a = (const float*)d_in[12];
  const float* ln2_b = (const float*)d_in[13];
  float* out = (float*)d_out;

  // Workspace partition (256B-aligned slices)
  char* p = (char*)d_ws;
  auto alloc = [&](size_t bytes) -> void* {
    void* r = (void*)p;
    p += (bytes + 255) & ~(size_t)255;
    return r;
  };
  _Float16* wqh   = (_Float16*)alloc((size_t)kNE * 3 * kNE * 2);
  _Float16* woh   = (_Float16*)alloc((size_t)kNE * kNE * 2);
  _Float16* w1h   = (_Float16*)alloc((size_t)kNE * kNHID * 2);
  _Float16* w2h   = (_Float16*)alloc((size_t)kNHID * kNE * 2);
  _Float16* xh    = (_Float16*)alloc((size_t)kTOK * kNE * 2);
  _Float16* qkvh  = (_Float16*)alloc((size_t)kTOK * 3 * kNE * 2);
  _Float16* kTb   = (_Float16*)alloc((size_t)kTOK * kNE * 2);
  _Float16* yh    = (_Float16*)alloc((size_t)kTOK * kNE * 2);
  float*    aout  = (float*)   alloc((size_t)kTOK * kNE * 4);
  float*    x1    = (float*)   alloc((size_t)kTOK * kNE * 4);
  _Float16* x1h   = (_Float16*)alloc((size_t)kTOK * kNE * 2);
  _Float16* hbufh = (_Float16*)alloc((size_t)kTOK * kNHID * 2);
  float*    h2    = (float*)   alloc((size_t)kTOK * kNE * 4);

  const dim3 gblk(128);
  auto cvt = [&](const float* s, _Float16* d, int n) {
    f32_to_f16_kernel<<<dim3(n / 1024), dim3(256), 0, stream>>>(s, d, n);
  };

  // 0) one-time f32->f16 conversions (weights stay resident in the 192MB L2)
  cvt(qkv_w, wqh, kNE * 3 * kNE);
  cvt(o_w,   woh, kNE * kNE);
  cvt(w1,    w1h, kNE * kNHID);
  cvt(w2,    w2h, kNHID * kNE);
  cvt(x,     xh,  kTOK * kNE);

  // 1) qkv = x @ qkv_w + qkv_b   -> f16            (8192 x 3072, K=1024)
  wmma_gemm_f16<0, 1><<<dim3(3 * kNE / 128, kTOK / 128), gblk, 0, stream>>>(
      xh, wqh, qkv_b, nullptr, qkvh, kTOK, 3 * kNE, kNE);

  // 1b) kT[b][h][d][s] = transpose of K section
  k_transpose_kernel<<<dim3((kSL / 64) * kBS * kNH), dim3(256), 0, stream>>>(qkvh, kTb);

  // 2) y = causal flash attention(q,k,v) -> f16
  flash_attn_kernel<<<dim3(kBS * kNH * (kSL / 16)), dim3(32), 0, stream>>>(
      qkvh, kTb, yh);

  // 3) attn_out = y @ o_w + o_b  -> f32            (8192 x 1024, K=1024)
  wmma_gemm_f16<0, 0><<<dim3(kNE / 128, kTOK / 128), gblk, 0, stream>>>(
      yh, woh, o_b, aout, nullptr, kTOK, kNE, kNE);

  // 4) x1 = LayerNorm(x + attn_out) -> f32 + f16
  add_ln_kernel<1><<<dim3(kTOK), dim3(256), 0, stream>>>(x, aout, ln1_a, ln1_b,
                                                         x1, x1h);

  // 5) h = relu(x1 @ w1 + b1) -> f16               (8192 x 4096, K=1024)
  wmma_gemm_f16<1, 1><<<dim3(kNHID / 128, kTOK / 128), gblk, 0, stream>>>(
      x1h, w1h, b1, nullptr, hbufh, kTOK, kNHID, kNE);

  // 6) h2 = h @ w2 + b2 -> f32                     (8192 x 1024, K=4096)
  wmma_gemm_f16<0, 0><<<dim3(kNE / 128, kTOK / 128), gblk, 0, stream>>>(
      hbufh, w2h, b2, h2, nullptr, kTOK, kNE, kNHID);

  // 7) out = LayerNorm(x1 + h2) -> f32 (d_out)
  add_ln_kernel<0><<<dim3(kTOK), dim3(256), 0, stream>>>(x1, h2, ln2_a, ln2_b,
                                                         out, nullptr);
}
